// GNN_30305289241228
// MI455X (gfx1250) — compile-verified, matrix operand order checked by reading, not run on hardware
//
#include <hip/hip_runtime.h>
#include <math.h>

// ---------------------------------------------------------------------------
// Problem constants (from the reference)
// ---------------------------------------------------------------------------
#define NROWS 500000
#define KDIM  100
#define KPAD  128     // K padded to 4 WMMA k-steps of 32
#define HDIM  256
#define RPB   128     // rows per block (8 waves x 16 rows)
#define TPB   256     // threads per block

typedef __attribute__((ext_vector_type(16))) _Float16 v16h;
typedef __attribute__((ext_vector_type(8)))  float    v8f;

union FragU { v16h v; uint4 q[2]; };

// packed B-fragment sizes (halves)
#define SZ1 (4 * 16 * 32 * 16)   // W1: 4 k-tiles x 16 n-tiles  = 32768 halves
#define SZ2 (8 * 16 * 32 * 16)   // W2/W3: 8 k-tiles x 16 n-tiles = 65536 halves
#define WS_HALVES (SZ1 + 2 * SZ2)

__device__ __forceinline__ float fast_tanh(float x) {
#if __has_builtin(__builtin_amdgcn_tanhf)
  return __builtin_amdgcn_tanhf(x);
#elif __has_builtin(__builtin_amdgcn_tanh_f32)
  return __builtin_amdgcn_tanh_f32(x);
#else
  return tanhf(x);
#endif
}

// ---------------------------------------------------------------------------
// Kernel 0: pack W1/W2/W3 (f32 row-major [k][n]) into f16 WMMA B fragments.
// Layout: half index = ((kt*16 + nt)*32 + lane)*16 + j
//   n    = nt*16 + (lane & 15)
//   koff = (j & 7) + ((lane >> 4) << 3) + ((j >> 3) << 4)   (16-bit frag pattern)
//   k    = kt*32 + koff    (zero-padded past kmax for W1)
// ---------------------------------------------------------------------------
__global__ __launch_bounds__(256) void pack_weights(
    const float* __restrict__ W1, const float* __restrict__ W2,
    const float* __restrict__ W3, _Float16* __restrict__ ws) {
  int gid = blockIdx.x * 256 + threadIdx.x;
  if (gid >= WS_HALVES) return;
  const float* W;
  int off, kmax;
  if (gid < SZ1)            { W = W1; off = gid;              kmax = KDIM; }
  else if (gid < SZ1 + SZ2) { W = W2; off = gid - SZ1;        kmax = HDIM; }
  else                      { W = W3; off = gid - SZ1 - SZ2;  kmax = HDIM; }
  int j    = off & 15;
  int lane = (off >> 4) & 31;
  int tile = off >> 9;          // kt*16 + nt
  int nt   = tile & 15;
  int kt   = tile >> 4;
  int n    = nt * 16 + (lane & 15);
  int koff = (j & 7) + ((lane >> 4) << 3) + ((j >> 3) << 4);
  int k    = kt * 32 + koff;
  float v  = (k < kmax) ? W[k * HDIM + n] : 0.0f;
  ws[gid]  = (_Float16)v;
}

// ---------------------------------------------------------------------------
// Main fused MLP kernel. One wave owns 16 rows; all GEMMs via WMMA f16->f32.
// Single 64KB LDS buffer reused in place (A fragments hoisted to registers).
// ---------------------------------------------------------------------------
__global__ __launch_bounds__(TPB) void gnn_mlp(
    const float* __restrict__ e,          // (N*K) edge_attr
    const _Float16* __restrict__ wpack,   // packed W1|W2|W3 fragments
    const float* __restrict__ b1, const float* __restrict__ b2,
    const float* __restrict__ b3,
    const float* __restrict__ W4, const float* __restrict__ b4,
    float* __restrict__ out) {
  __shared__ alignas(16) _Float16 lds[RPB * HDIM];  // 64 KB; bufE aliases 32 KB
  _Float16* bufE = lds;                   // [128][KPAD]  (layer-1 input)
  _Float16* bufH = lds;                   // [128][HDIM]  (activations, in place)

  const int t    = threadIdx.x;
  const int lane = t & 31;
  const int w    = t >> 5;
  const int R0   = blockIdx.x * RPB;
  const int mrow = lane & 15;   // A-frag row / C-tile column within 16
  const int hi   = lane >> 4;   // lane-group (adds +8 to K / selects M+8)
  const int rowbase = w * 16;

  // ---- stage edge_attr -> LDS as f16, pad K 100 -> 128 -------------------
  // 128 rows x 100 floats = 12800 contiguous floats; KDIM%4==0 so a float4
  // never crosses a row boundary -> vector loads + b64 LDS stores.
  {
    const float* eb = e + (size_t)R0 * KDIM;
    const int limit = (NROWS - R0 < RPB ? NROWS - R0 : RPB) * KDIM;
    #pragma unroll
    for (int i = 0; i < 13; ++i) {                 // ceil(3200/256)
      int idx4 = t + i * TPB;
      if (idx4 < (RPB * KDIM) / 4) {
        int fo = idx4 * 4;
        float4 v = make_float4(0.f, 0.f, 0.f, 0.f);
        if (fo < limit) v = *(const float4*)(eb + fo);
        int r = fo / KDIM, k = fo - r * KDIM;
        union { _Float16 h[4]; uint2 u; } pk;
        pk.h[0] = (_Float16)v.x; pk.h[1] = (_Float16)v.y;
        pk.h[2] = (_Float16)v.z; pk.h[3] = (_Float16)v.w;
        *(uint2*)(bufE + r * KPAD + k) = pk.u;
      }
    }
    for (int idx = t; idx < RPB * (KPAD - KDIM) / 4; idx += TPB) {
      int r = idx / ((KPAD - KDIM) / 4), k = KDIM + 4 * (idx - r * ((KPAD - KDIM) / 4));
      union { _Float16 h[4]; uint2 u; } z;
      z.h[0] = z.h[1] = z.h[2] = z.h[3] = (_Float16)0.0f;
      *(uint2*)(bufE + r * KPAD + k) = z.u;
    }
  }
  __syncthreads();

  const _Float16* B1 = wpack;
  const _Float16* B2 = wpack + SZ1;
  const _Float16* B3 = wpack + SZ1 + SZ2;

  // ---- Layer 1: (16 x 128) @ (128 x 256) ---------------------------------
  FragU a1[4];
  {
    const _Float16* abase = bufE + (rowbase + mrow) * KPAD + hi * 8;
    #pragma unroll
    for (int kt = 0; kt < 4; ++kt) {
      a1[kt].q[0] = *(const uint4*)(abase + kt * 32);       // K {0..7}+off
      a1[kt].q[1] = *(const uint4*)(abase + kt * 32 + 16);  // K {16..23}+off
    }
  }
  __syncthreads();   // all waves captured e-frags before bufH overwrites bufE

  #pragma unroll 4
  for (int nt = 0; nt < 16; ++nt) {
    float bias = b1[nt * 16 + mrow];
    v8f acc = {bias, bias, bias, bias, bias, bias, bias, bias};
    #pragma unroll
    for (int kt = 0; kt < 4; ++kt) {
      v16h bf = *(const v16h*)(B1 + ((size_t)(kt * 16 + nt) * 32 + lane) * 16);
      acc = __builtin_amdgcn_wmma_f32_16x16x32_f16(
          false, a1[kt].v, false, bf, (short)0, acc, false, false);
    }
    #pragma unroll
    for (int g = 0; g < 8; ++g) {
      float hv = fast_tanh(acc[g]);
      bufH[(rowbase + g + hi * 8) * HDIM + nt * 16 + mrow] = (_Float16)hv;
    }
  }

  // ---- Layers 2 & 3: (16 x 256) @ (256 x 256), in-place in bufH ----------
  for (int L = 0; L < 2; ++L) {
    const _Float16* Bp = (L == 0) ? B2 : B3;
    const float*    bp = (L == 0) ? b2 : b3;
    FragU a[8];
    const _Float16* abase = bufH + (rowbase + mrow) * HDIM + hi * 8;
    #pragma unroll
    for (int kt = 0; kt < 8; ++kt) {        // whole A tile -> 64 VGPRs
      a[kt].q[0] = *(const uint4*)(abase + kt * 32);
      a[kt].q[1] = *(const uint4*)(abase + kt * 32 + 16);
    }
    // DS ops are in-order per wave: loads above observe memory before the
    // stores below rewrite it; no cross-wave sharing -> no barrier needed.
    #pragma unroll 4
    for (int nt = 0; nt < 16; ++nt) {
      float bias = bp[nt * 16 + mrow];
      v8f acc = {bias, bias, bias, bias, bias, bias, bias, bias};
      #pragma unroll
      for (int kt = 0; kt < 8; ++kt) {
        v16h bf = *(const v16h*)(Bp + ((size_t)(kt * 16 + nt) * 32 + lane) * 16);
        acc = __builtin_amdgcn_wmma_f32_16x16x32_f16(
            false, a[kt].v, false, bf, (short)0, acc, false, false);
      }
      #pragma unroll
      for (int g = 0; g < 8; ++g) {
        float hv = fast_tanh(acc[g]);
        bufH[(rowbase + g + hi * 8) * HDIM + nt * 16 + mrow] = (_Float16)hv;
      }
    }
  }

  // ---- Layer 4: (16 x 256) @ (256 x 1) + sigmoid -------------------------
  {
    int r = rowbase + mrow;                       // local row; lane pairs split K
    const _Float16* hrow = bufH + r * HDIM + hi * 128;
    const float*    w4   = W4 + hi * 128;
    float sum = 0.0f;
    #pragma unroll 4
    for (int k = 0; k < 128; k += 8) {
      union { uint4 q; _Float16 h[8]; } u;
      u.q = *(const uint4*)(hrow + k);
      float4 w0 = *(const float4*)(w4 + k);
      float4 w1 = *(const float4*)(w4 + k + 4);
      sum += (float)u.h[0] * w0.x + (float)u.h[1] * w0.y +
             (float)u.h[2] * w0.z + (float)u.h[3] * w0.w +
             (float)u.h[4] * w1.x + (float)u.h[5] * w1.y +
             (float)u.h[6] * w1.z + (float)u.h[7] * w1.w;
    }
    sum += __shfl_xor(sum, 16, 32);               // combine K halves
    int grow = R0 + r;
    if (hi == 0 && grow < NROWS) {
      float z = sum + b4[0];
      out[grow] = 1.0f / (1.0f + __expf(-z));
    }
  }
}

// ---------------------------------------------------------------------------
// Launch
// ---------------------------------------------------------------------------
extern "C" void kernel_launch(void* const* d_in, const int* in_sizes, int n_in,
                              void* d_out, int out_size, void* d_ws, size_t ws_size,
                              hipStream_t stream) {
  // inputs: 0:x 1:edge_index 2:edge_attr 3:W1 4:b1 5:W2 6:b2 7:W3 8:b3 9:W4 10:b4
  const float* e  = (const float*)d_in[2];
  const float* W1 = (const float*)d_in[3];
  const float* b1 = (const float*)d_in[4];
  const float* W2 = (const float*)d_in[5];
  const float* b2 = (const float*)d_in[6];
  const float* W3 = (const float*)d_in[7];
  const float* b3 = (const float*)d_in[8];
  const float* W4 = (const float*)d_in[9];
  const float* b4 = (const float*)d_in[10];
  float* out = (float*)d_out;
  _Float16* wpack = (_Float16*)d_ws;   // needs WS_HALVES*2 = 320 KB of scratch

  int packBlocks = (WS_HALVES + 255) / 256;        // 640
  pack_weights<<<packBlocks, 256, 0, stream>>>(W1, W2, W3, wpack);

  int blocks = (NROWS + RPB - 1) / RPB;            // 3907
  gnn_mlp<<<blocks, TPB, 0, stream>>>(e, wpack, b1, b2, b3, W4, b4, out);
}
